// TSF_attention_13898514170133
// MI455X (gfx1250) — matrix operand order repaired and checked by reading, not compiled, not run on hardware
//
#include <hip/hip_runtime.h>

typedef __attribute__((ext_vector_type(8)))  float    v8f;
typedef __attribute__((ext_vector_type(8)))  __bf16   v8bf;
typedef __attribute__((ext_vector_type(16))) __bf16   v16bf;
typedef __attribute__((ext_vector_type(4)))  unsigned v4u;
typedef __attribute__((ext_vector_type(8)))  unsigned v8u;

#define SDIM 65536   // 16*64*64 spatial positions
#define CDIM 384

__device__ __forceinline__ v8f wmma_bf16(v16bf a, v16bf b, v8f c) {
  return __builtin_amdgcn_wmma_f32_16x16x32_bf16(false, a, false, b, (short)0, c,
                                                 false, false);
}
__device__ __forceinline__ float gelu_exact(float x) {
  return 0.5f * x * (1.0f + erff(x * 0.70710678118654752f));
}

struct HCPtrs { const float* fcw; const float* fcb; const float* fcbw; const float* fcbb; };

// Shared tile memory: single object => LDS offset 0 for A (needed by TDM/async asm)
struct TileSmem {
  __bf16 A[128 * 32];   // offset 0, 8 KB : A chunk, row-major [m][k]
  __bf16 B[64 * 32];    // offset 8192, 4 KB : B chunk, transposed [n][k]
};

// ---------------------------------------------------------------------------
// Small-region layout in workspace (float indices into sm):
//  [0,48) z codes   [48,96) zb codes   [96,100) mix wgt0..2, 1/sum
//  [128,512) cattn
//  [512,10880) dwk1   [10880,11264) dwb1
//  [11264,21632) dwk2  [21632,22016) dwb2
//  [22016,23552) bias pw1 blk1  [23552,23936) bias pw2 blk1
//  [23936,25472) bias pw1 res1  [25472,25856) bias pw2 res1
// ---------------------------------------------------------------------------

__global__ void k_style(const float* __restrict__ s,
                        const float* __restrict__ fcw_w, const float* __restrict__ fcw_b,
                        HCPtrs h0, HCPtrs h1, HCPtrs h2, HCPtrs h3, HCPtrs h4, HCPtrs h5,
                        const float* __restrict__ dw1_param, const float* __restrict__ dw1_b,
                        const float* __restrict__ dw2_param, const float* __restrict__ dw2_b,
                        const float* __restrict__ pw11_b, const float* __restrict__ pw21_b,
                        const float* __restrict__ pw12_b, const float* __restrict__ pw22_b,
                        const float* __restrict__ nsw, const float* __restrict__ nsb,
                        const float* __restrict__ v_w,
                        const float* __restrict__ proj_w, const float* __restrict__ proj_b,
                        float* __restrict__ sm)
{
  __shared__ float zsh[48], zbsh[48], snsh[6], vvec[384];
  int tid = threadIdx.x;
  HCPtrs hc[6] = {h0, h1, h2, h3, h4, h5};
  if (tid < 48) {
    int j = tid >> 3, r = tid & 7;
    float a = hc[j].fcb[r], b = hc[j].fcbb[r];
    for (int k = 0; k < 6; ++k) {
      a += hc[j].fcw[r * 6 + k] * s[k];
      b += hc[j].fcbw[r * 6 + k] * s[k];
    }
    zsh[tid] = a; zbsh[tid] = b;
    sm[tid] = a; sm[48 + tid] = b;
  }
  if (tid == 64) {                         // ln_last over style (eps 1e-6)
    float u = 0.f; for (int k = 0; k < 6; ++k) u += s[k]; u *= (1.f / 6.f);
    float v = 0.f; for (int k = 0; k < 6; ++k) { float d = s[k] - u; v += d * d; }
    v *= (1.f / 6.f);
    float r = rsqrtf(v + 1e-6f);
    for (int k = 0; k < 6; ++k) snsh[k] = nsw[k] * (s[k] - u) * r + nsb[k];
  }
  if (tid == 65) {                         // mixing weights (softmax + 1e-5)
    float lg[3];
    for (int j = 0; j < 3; ++j) {
      float a = fcw_b[j];
      for (int k = 0; k < 6; ++k) a += fcw_w[j * 6 + k] * s[k];
      lg[j] = a;
    }
    float mx = fmaxf(lg[0], fmaxf(lg[1], lg[2]));
    float e0 = expf(lg[0] - mx), e1 = expf(lg[1] - mx), e2 = expf(lg[2] - mx);
    float den = e0 + e1 + e2;
    float g0 = s[0] * (e0 / den + 1e-5f);
    float g1 = s[1] * (e1 / den + 1e-5f);
    float g2 = s[2] * (e2 / den + 1e-5f);
    sm[96] = g0; sm[97] = g1; sm[98] = g2; sm[99] = 1.f / (g0 + g1 + g2);
  }
  __syncthreads();
  for (int idx = tid; idx < 10368; idx += 256) {   // depthwise hyper-kernels
    float a = 0.f, b = 0.f;
    for (int k = 0; k < 8; ++k) {
      a += dw1_param[idx * 8 + k] * zsh[k];
      b += dw2_param[idx * 8 + k] * zsh[24 + k];
    }
    sm[512 + idx] = a; sm[11264 + idx] = b;
  }
  for (int c = tid; c < 384; c += 256) {           // depthwise biases
    float a = 0.f, b = 0.f;
    for (int k = 0; k < 8; ++k) {
      a += dw1_b[c * 8 + k] * zbsh[k];
      b += dw2_b[c * 8 + k] * zbsh[24 + k];
    }
    sm[10880 + c] = a; sm[21632 + c] = b;
  }
  for (int o = tid; o < 1536; o += 256) {          // pw1 biases
    float a = 0.f, b = 0.f;
    for (int k = 0; k < 8; ++k) {
      a += pw11_b[o * 8 + k] * zbsh[8 + k];
      b += pw12_b[o * 8 + k] * zbsh[32 + k];
    }
    sm[22016 + o] = a; sm[23936 + o] = b;
  }
  for (int o = tid; o < 384; o += 256) {           // pw2 biases
    float a = 0.f, b = 0.f;
    for (int k = 0; k < 8; ++k) {
      a += pw21_b[o * 8 + k] * zbsh[16 + k];
      b += pw22_b[o * 8 + k] * zbsh[40 + k];
    }
    sm[23552 + o] = a; sm[25472 + o] = b;
  }
  for (int c = tid; c < 384; c += 256) {           // v = v_w @ ln(s)
    float a = 0.f;
    for (int k = 0; k < 6; ++k) a += v_w[c * 6 + k] * snsh[k];
    vvec[c] = a;
  }
  __syncthreads();
  // cross-attn collapses to per-channel constant (softmax over 1 token == 1)
  for (int o = tid; o < 384; o += 256) {
    float a = proj_b[o];
    for (int c = 0; c < 384; ++c) a += proj_w[o * 384 + c] * vvec[c];
    sm[128 + o] = a;
  }
}

__global__ void k_makeW(const float* __restrict__ param, const float* __restrict__ z,
                        __bf16* __restrict__ W, int total) {
  int idx = blockIdx.x * 256 + threadIdx.x;
  if (idx >= total) return;
  const float* p = param + (size_t)idx * 8;
  float a = p[0] * z[0] + p[1] * z[1] + p[2] * z[2] + p[3] * z[3] +
            p[4] * z[4] + p[5] * z[5] + p[6] * z[6] + p[7] * z[7];
  W[idx] = (__bf16)a;
}

__global__ void k_makeWoc(const float* __restrict__ w, __bf16* __restrict__ Woc) {
  int idx = blockIdx.x * 256 + threadIdx.x;          // < 27*128*384
  int t = idx / 49152; int rem = idx - t * 49152;
  int o = rem / 384;   int c = rem - o * 384;
  Woc[idx] = (__bf16)w[(size_t)o * 10368 + c * 27 + t];
}

__global__ void k_mix(const float* __restrict__ zs, const float* __restrict__ sm,
                      float* __restrict__ out) {
  int idx = blockIdx.x * 256 + threadIdx.x;          // < 128*65536
  float w0 = sm[96], w1 = sm[97], w2 = sm[98], inv = sm[99];
  out[idx] = (zs[idx] * w0 + zs[idx + 8388608] * w1 + zs[idx + 16777216] * w2) * inv;
}

__global__ void k_dw(const float* __restrict__ x, const float* __restrict__ dwk,
                     const float* __restrict__ dwb, float* __restrict__ out) {
  int bi = blockIdx.x;                               // c*256 + chunk
  int c = bi >> 8;
  int p = ((bi & 255) << 8) + threadIdx.x;
  int d = p >> 12, h = (p >> 6) & 63, wx = p & 63;
  const float* xc = x + (size_t)c * SDIM;
  const float* kc = dwk + c * 27;
  float acc = dwb[c];
  #pragma unroll
  for (int t = 0; t < 27; ++t) {
    int dz = t / 9 - 1, dy = (t / 3) % 3 - 1, dx = t % 3 - 1;
    int dd = d + dz, hh = h + dy, ww = wx + dx;
    if ((unsigned)dd < 16u && (unsigned)hh < 64u && (unsigned)ww < 64u)
      acc += kc[t] * xc[(dd << 12) + (hh << 6) + ww];
  }
  out[(size_t)c * SDIM + p] = acc;
}

__global__ void k_lnstats(const float* __restrict__ x, float* __restrict__ mean,
                          float* __restrict__ rstd) {
  int p = blockIdx.x * 256 + threadIdx.x;
  float s = 0.f, s2 = 0.f;
  for (int c = 0; c < CDIM; ++c) {
    float v = x[(size_t)c * SDIM + p];
    s += v; s2 += v * v;
  }
  float u = s * (1.f / CDIM);
  float var = s2 * (1.f / CDIM) - u * u;
  mean[p] = u; rstd[p] = rsqrtf(var + 1e-6f);
}

__global__ void k_lnnorm(const float* __restrict__ x, const float* __restrict__ mean,
                         const float* __restrict__ rstd, const float* __restrict__ gw,
                         const float* __restrict__ gb, __bf16* __restrict__ y) {
  size_t idx = (size_t)blockIdx.x * 256 + threadIdx.x;   // < 384*65536
  int c = (int)(idx >> 16), p = (int)(idx & 65535);
  y[idx] = (__bf16)(gw[c] * (x[idx] - mean[p]) * rstd[p] + gb[c]);
}

// C(M x 65536) = A(M x K) * X(K x 65536), bf16 WMMA f32-acc.
// Block tile M=128 x N=64, 8 waves, 4 WMMA / wave / K-step.
// A chunk staged by the Tensor Data Mover (wave 0 issues tensor_load_to_lds).
__global__ void __launch_bounds__(256) k_gemm(
    const __bf16* __restrict__ A, const __bf16* __restrict__ X,
    const float* __restrict__ bias, int K, int mode,
    const float* __restrict__ resid, const float* __restrict__ gamma,
    const float* __restrict__ addvec,
    float* __restrict__ outf, __bf16* __restrict__ outh)
{
  __shared__ TileSmem smem;
  int tid = threadIdx.x, lane = tid & 31, wv = tid >> 5;
  int n0 = blockIdx.x * 64;
  int m0 = blockIdx.y * 128;
  v8f z8 = {0, 0, 0, 0, 0, 0, 0, 0};
  v8f accs[4] = {z8, z8, z8, z8};

  int bk = tid >> 3, bn = (tid & 7) * 8;           // B coop load: 8 halfs/thread
  int mrow = 16 * wv + (lane & 15);
  int kb  = (lane >> 4) * 16;                       // A frag: contiguous K run kb..kb+15
  int kb2 = (lane >> 4) * 16;                       // B frag: contiguous K run
  int ncol = lane & 15;

  // TDM descriptor group 1 (invariant): data_size=8B, tile 8x128 units,
  // tensor dims/stride sized to the A panel row pitch (K/4 units of 8B).
  unsigned ku = (unsigned)(K >> 2);
  v8u g1;
  g1[0] = 3u << 16;                         // workgroup_mask=0, data_size=3 (8B)
  g1[1] = (ku & 0xFFFFu) << 16;             // tensor_dim0[15:0]
  g1[2] = (ku >> 16) | (128u << 16);        // tensor_dim0[31:16] | tensor_dim1[15:0]
  g1[3] = 8u << 16;                         // tensor_dim1 hi=0 | tile_dim0=8
  g1[4] = 128u;                             // tile_dim1=128, tile_dim2=0
  g1[5] = ku;                               // tensor_dim0_stride[31:0]
  g1[6] = 0u;                               // stride0 hi | stride1 lo
  g1[7] = 0u;

  for (int k0 = 0; k0 < K; k0 += 32) {
    const __bf16* bsrc = X + (size_t)(k0 + bk) * SDIM + n0 + bn;
    v8bf bv = *(const v8bf*)bsrc;
    if (k0 + 32 < K)
      __builtin_prefetch(X + (size_t)(k0 + 32 + bk) * SDIM + n0 + bn, 0, 0);
    __syncthreads();                        // previous iteration reads done
    if (wv == 0) {                          // TDM: A chunk (128 x 32 bf16) -> LDS 0
      unsigned long long ga =
          (unsigned long long)(const void*)(A + (size_t)m0 * K + k0);
      v4u g0;
      g0[0] = 1u;                           // count=1 valid descriptor
      g0[1] = 0u;                           // lds_addr = 0 (smem.A)
      g0[2] = (unsigned)(ga & 0xFFFFFFFFull);
      g0[3] = (unsigned)((ga >> 32) & 0x1FFFFFFull) | (2u << 30);  // type=2
      asm volatile("tensor_load_to_lds %0, %1" :: "s"(g0), "s"(g1) : "memory");
    }
    #pragma unroll
    for (int j = 0; j < 8; ++j) smem.B[(bn + j) * 32 + bk] = bv[j];
    if (wv == 0) __builtin_amdgcn_s_wait_tensorcnt(0);
    __syncthreads();
    v8bf a0 = *(const v8bf*)&smem.A[mrow * 32 + kb];
    v8bf a1 = *(const v8bf*)&smem.A[mrow * 32 + kb + 8];
    v16bf a = __builtin_shufflevector(a0, a1, 0, 1, 2, 3, 4, 5, 6, 7,
                                      8, 9, 10, 11, 12, 13, 14, 15);
    #pragma unroll
    for (int nt = 0; nt < 4; ++nt) {
      v16bf b = *(const v16bf*)&smem.B[(nt * 16 + ncol) * 32 + kb2];
      accs[nt] = wmma_bf16(a, b, accs[nt]);
    }
  }
  int lr = lane & 15, hi = lane >> 4;
  #pragma unroll
  for (int nt = 0; nt < 4; ++nt) {
    #pragma unroll
    for (int r = 0; r < 8; ++r) {
      int row = m0 + 16 * wv + r + hi * 8;
      int col = n0 + nt * 16 + lr;
      size_t oidx = (size_t)row * SDIM + col;
      float v = accs[nt][r] + bias[row];
      if (mode == 0) {
        outh[oidx] = (__bf16)gelu_exact(v);
      } else {
        float o = resid[oidx] + gamma[row] * v;
        if (addvec) o += addvec[row];
        outf[oidx] = o;
      }
    }
  }
}

// implicit-GEMM 3x3x3 conv 384->128, WMMA; A tiles staged with async-to-LDS.
__global__ void __launch_bounds__(256) k_outconv(
    const __bf16* __restrict__ Woc, const __bf16* __restrict__ xln,
    const float* __restrict__ ob, float* __restrict__ out)
{
  __shared__ TileSmem smem;
  int tid = threadIdx.x, lane = tid & 31, wv = tid >> 5;
  int n0 = blockIdx.x * 64;
  v8f z8 = {0, 0, 0, 0, 0, 0, 0, 0};
  v8f accs[4] = {z8, z8, z8, z8};
  int am = tid >> 1, akc = (tid & 1) * 16;
  int bk = tid >> 3, bn = (tid & 7) * 8;
  int mrow = 16 * wv + (lane & 15);
  int kb  = (lane >> 4) * 16;
  int kb2 = (lane >> 4) * 16;
  int ncol = lane & 15;
  unsigned aoff = (unsigned)((am * 32 + akc) * 2);   // LDS byte offset in smem.A

  for (int t = 0; t < 27; ++t) {
    int dz = t / 9 - 1, dy = (t / 3) % 3 - 1, dx = t % 3 - 1;
    int shift = (dz << 12) + (dy << 6) + dx;
    const __bf16* At = Woc + (size_t)t * 49152;
    for (int kc = 0; kc < 384; kc += 32) {
      __bf16 bv[8];
      #pragma unroll
      for (int j = 0; j < 8; ++j) {
        int p = n0 + bn + j;
        int d = (p >> 12) + dz, h = ((p >> 6) & 63) + dy, wx = (p & 63) + dx;
        __bf16 v = (__bf16)0.0f;
        if ((unsigned)d < 16u && (unsigned)h < 64u && (unsigned)wx < 64u)
          v = xln[(size_t)(kc + bk) * SDIM + p + shift];
        bv[j] = v;
      }
      __syncthreads();
      {   // async copy: 32 bytes of one A row per thread into smem.A
        unsigned long long ga =
            (unsigned long long)(const void*)(At + (size_t)am * 384 + kc + akc);
        asm volatile("global_load_async_to_lds_b128 %0, %1, off"
                     :: "v"(aoff), "v"(ga) : "memory");
        asm volatile("global_load_async_to_lds_b128 %0, %1, off offset:16"
                     :: "v"(aoff), "v"(ga) : "memory");
      }
      #pragma unroll
      for (int j = 0; j < 8; ++j) smem.B[(bn + j) * 32 + bk] = bv[j];
      asm volatile("s_wait_asynccnt 0x0" ::: "memory");
      __syncthreads();
      v8bf a0 = *(const v8bf*)&smem.A[mrow * 32 + kb];
      v8bf a1 = *(const v8bf*)&smem.A[mrow * 32 + kb + 8];
      v16bf a = __builtin_shufflevector(a0, a1, 0, 1, 2, 3, 4, 5, 6, 7,
                                        8, 9, 10, 11, 12, 13, 14, 15);
      #pragma unroll
      for (int nt = 0; nt < 4; ++nt) {
        v16bf b = *(const v16bf*)&smem.B[(nt * 16 + ncol) * 32 + kb2];
        accs[nt] = wmma_bf16(a, b, accs[nt]);
      }
    }
  }
  int lr = lane & 15, hi = lane >> 4;
  #pragma unroll
  for (int nt = 0; nt < 4; ++nt) {
    #pragma unroll
    for (int r = 0; r < 8; ++r) {
      int row = 16 * wv + r + hi * 8;
      int col = n0 + nt * 16 + lr;
      size_t oidx = (size_t)row * SDIM + col;
      out[oidx] = out[oidx] + accs[nt][r] + ob[row];
    }
  }
}

extern "C" void kernel_launch(void* const* d_in, const int* in_sizes, int n_in,
                              void* d_out, int out_size, void* d_ws, size_t ws_size,
                              hipStream_t stream) {
  (void)in_sizes; (void)n_in; (void)out_size; (void)ws_size;
  int i = 0;
  const float* zs    = (const float*)d_in[i++];
  const float* s     = (const float*)d_in[i++];
  const float* fcw_w = (const float*)d_in[i++];
  const float* fcw_b = (const float*)d_in[i++];

#define GET_HCONV(H, P, B)                                                     \
  H.fcw  = (const float*)d_in[i++]; H.fcb  = (const float*)d_in[i++];          \
  P      = (const float*)d_in[i++];                                            \
  H.fcbw = (const float*)d_in[i++]; H.fcbb = (const float*)d_in[i++];          \
  B      = (const float*)d_in[i++];

  HCPtrs b1dw, b1p1, b1p2, r1dw, r1p1, r1p2;
  const float *b1dw_param, *b1dw_b, *b1p1_param, *b1p1_b, *b1p2_param, *b1p2_b;
  const float *r1dw_param, *r1dw_b, *r1p1_param, *r1p1_b, *r1p2_param, *r1p2_b;

  GET_HCONV(b1dw, b1dw_param, b1dw_b);
  const float* b1_nw = (const float*)d_in[i++];
  const float* b1_nb = (const float*)d_in[i++];
  GET_HCONV(b1p1, b1p1_param, b1p1_b);
  GET_HCONV(b1p2, b1p2_param, b1p2_b);
  const float* b1_gamma = (const float*)d_in[i++];

  const float* a_nw    = (const float*)d_in[i++]; (void)a_nw;   // dead: q path
  const float* a_nb    = (const float*)d_in[i++]; (void)a_nb;
  const float* ns_w    = (const float*)d_in[i++];
  const float* ns_b    = (const float*)d_in[i++];
  const float* q_w     = (const float*)d_in[i++]; (void)q_w;    // dead: softmax(1)=1
  const float* k_w     = (const float*)d_in[i++]; (void)k_w;
  const float* v_w     = (const float*)d_in[i++];
  const float* proj_w  = (const float*)d_in[i++];
  const float* proj_b  = (const float*)d_in[i++];

  GET_HCONV(r1dw, r1dw_param, r1dw_b);
  const float* r1_nw = (const float*)d_in[i++];
  const float* r1_nb = (const float*)d_in[i++];
  GET_HCONV(r1p1, r1p1_param, r1p1_b);
  GET_HCONV(r1p2, r1p2_param, r1p2_b);
  const float* r1_gamma = (const float*)d_in[i++];

  const float* out_ln_w   = (const float*)d_in[i++];
  const float* out_ln_b   = (const float*)d_in[i++];
  const float* out_conv_w = (const float*)d_in[i++];
  const float* out_conv_b = (const float*)d_in[i++];
#undef GET_HCONV

  // ---- workspace carve-up (~344 MB total) ----
  char* ws = (char*)d_ws;
  float* sm = (float*)ws;
  size_t off = 32768ull * 4;                                 // small region
  __bf16* W11 = (__bf16*)(ws + off); off += 589824ull * 2;   // 1536x384
  __bf16* W21 = (__bf16*)(ws + off); off += 589824ull * 2;   // 384x1536
  __bf16* W12 = (__bf16*)(ws + off); off += 589824ull * 2;
  __bf16* W22 = (__bf16*)(ws + off); off += 589824ull * 2;
  __bf16* Woc = (__bf16*)(ws + off); off += 1327104ull * 2;  // 27x128x384
  float*  meanb = (float*)(ws + off); off += 65536ull * 4;
  float*  rstdb = (float*)(ws + off); off += 65536ull * 4;
  __bf16* xln  = (__bf16*)(ws + off); off += 25165824ull * 2;   // (384,S) bf16
  float*  xcur = (float*)(ws + off);  off += 25165824ull * 4;   // (384,S) f32
  __bf16* Hbuf = (__bf16*)(ws + off); off += 100663296ull * 2;  // (1536,S) bf16
  float*  tdw  = (float*)Hbuf;   // alias: dw-conv scratch dead before Hbuf live

  float* dout = (float*)d_out;

  k_style<<<1, 256, 0, stream>>>(s, fcw_w, fcw_b,
      b1dw, b1p1, b1p2, r1dw, r1p1, r1p2,
      b1dw_param, b1dw_b, r1dw_param, r1dw_b,
      b1p1_b, b1p2_b, r1p1_b, r1p2_b,
      ns_w, ns_b, v_w, proj_w, proj_b, sm);
  k_makeW<<<2304, 256, 0, stream>>>(b1p1_param, sm + 8,  W11, 589824);
  k_makeW<<<2304, 256, 0, stream>>>(b1p2_param, sm + 16, W21, 589824);
  k_makeW<<<2304, 256, 0, stream>>>(r1p1_param, sm + 32, W12, 589824);
  k_makeW<<<2304, 256, 0, stream>>>(r1p2_param, sm + 40, W22, 589824);
  k_makeWoc<<<5184, 256, 0, stream>>>(out_conv_w, Woc);
  k_mix<<<32768, 256, 0, stream>>>(zs, sm, dout);

  // hyper_block 1 (input = zs)
  k_dw<<<98304, 256, 0, stream>>>(zs, sm + 512, sm + 10880, tdw);
  k_lnstats<<<256, 256, 0, stream>>>(tdw, meanb, rstdb);
  k_lnnorm<<<98304, 256, 0, stream>>>(tdw, meanb, rstdb, b1_nw, b1_nb, xln);
  k_gemm<<<dim3(1024, 12), 256, 0, stream>>>(W11, xln, sm + 22016, 384, 0,
      nullptr, nullptr, nullptr, nullptr, Hbuf);
  k_gemm<<<dim3(1024, 3), 256, 0, stream>>>(W21, Hbuf, sm + 23552, 1536, 1,
      zs, b1_gamma, sm + 128 /*cattn*/, xcur, nullptr);

  // hyper_block 2 (res1, input = xcur)
  k_dw<<<98304, 256, 0, stream>>>(xcur, sm + 11264, sm + 21632, tdw);
  k_lnstats<<<256, 256, 0, stream>>>(tdw, meanb, rstdb);
  k_lnnorm<<<98304, 256, 0, stream>>>(tdw, meanb, rstdb, r1_nw, r1_nb, xln);
  k_gemm<<<dim3(1024, 12), 256, 0, stream>>>(W12, xln, sm + 23936, 384, 0,
      nullptr, nullptr, nullptr, nullptr, Hbuf);
  k_gemm<<<dim3(1024, 3), 256, 0, stream>>>(W22, Hbuf, sm + 25472, 1536, 1,
      xcur, r1_gamma, nullptr, xcur, nullptr);

  // final LN + 3x3x3 conv 384->128 + add into mix
  k_lnstats<<<256, 256, 0, stream>>>(xcur, meanb, rstdb);
  k_lnnorm<<<98304, 256, 0, stream>>>(xcur, meanb, rstdb, out_ln_w, out_ln_b, xln);
  k_outconv<<<1024, 256, 0, stream>>>(Woc, xln, out_conv_b, dout);
}
